// MixtureOfExpertsMLPModule_62466004353423
// MI455X (gfx1250) — compile-verified, matrix operand order checked by reading, not run on hardware
//
#include <hip/hip_runtime.h>

typedef __attribute__((ext_vector_type(16))) __bf16 v16bf;
typedef __attribute__((ext_vector_type(8)))  __bf16 v8bf;
typedef __attribute__((ext_vector_type(8)))  float  v8f;

static constexpr int TOK = 8192;   // B*S
static constexpr int DIM = 1024;   // D
static constexpr int CHN = 256;    // CH
static constexpr int BM  = 32;     // tokens per block
static constexpr int KH  = 512;    // K half for split-K A staging

// ---------------- fp32 -> bf16 weight conversion ----------------
__global__ __launch_bounds__(256) void cvt_bf16_kernel(const float* __restrict__ src,
                                                       __bf16* __restrict__ dst, int n) {
  int i = blockIdx.x * blockDim.x + threadIdx.x;
  int stride = gridDim.x * blockDim.x;
  for (; i < n; i += stride) dst[i] = (__bf16)src[i];
}

// ---------------- input layernorm: one token per wave ----------------
__global__ __launch_bounds__(256) void xnorm_kernel(const float* __restrict__ x,
                                                    const float* __restrict__ g,
                                                    const float* __restrict__ b,
                                                    float* __restrict__ xn) {
  int token = blockIdx.x * 8 + (threadIdx.x >> 5);
  int lane  = threadIdx.x & 31;
  const float* row = x + (size_t)token * DIM;
  float s = 0.f, s2 = 0.f;
  for (int i = lane; i < DIM; i += 32) { float v = row[i]; s += v; s2 += v * v; }
#pragma unroll
  for (int m = 16; m >= 1; m >>= 1) { s += __shfl_xor(s, m, 32); s2 += __shfl_xor(s2, m, 32); }
  float mu   = s  * (1.f / DIM);
  float var  = s2 * (1.f / DIM) - mu * mu;
  float rstd = rsqrtf(var + 1e-5f);
  float* o = xn + (size_t)token * DIM;
  for (int i = lane; i < DIM; i += 32) o[i] = (row[i] - mu) * rstd * g[i] + b[i];
}

// WMMA fragment loader: 16 bf16 per lane as two contiguous 16B chunks
__device__ __forceinline__ v16bf load_frag(const __bf16* p) {
  v8bf lo = *(const v8bf*)p;
  v8bf hi = *(const v8bf*)(p + 16);
  v16bf f;
#pragma unroll
  for (int i = 0; i < 8; ++i) { f[i] = lo[i]; f[8 + i] = hi[i]; }
  return f;
}

// ---------------- fused per-node kernel (32 tokens / block) ----------------
__global__ __launch_bounds__(256) void node_kernel(
    const float* __restrict__ xnorm,
    const __bf16* __restrict__ w1,      // [CHN][DIM] bf16
    const float*  __restrict__ b1,      // [CHN]
    const float*  __restrict__ lng, const float* __restrict__ lnb,
    const __bf16* __restrict__ w2,      // [DIM][CHN] bf16
    const float*  __restrict__ rw,      // router_w[v]: [4][DIM]
    const float*  __restrict__ rb,      // router_b[v]: [4]
    const float* __restrict__ p0, const float* __restrict__ p1,
    const float* __restrict__ p2, const float* __restrict__ p3,
    int npred,
    float* __restrict__ outbuf,
    const float* __restrict__ x,
    const float* __restrict__ fc2_bias,
    int terminal)
{
  __shared__ __bf16 Ain[BM][KH + 8];    // node_in K-half tile, bf16 (padded)
  __shared__ float  wgt[BM][4];         // sigmoid gates
  __shared__ float  part[BM][8][2];     // per-wave LN partials (sum, sumsq)
  __shared__ float  musr[BM][2];        // per-token (mu, rstd)
  __shared__ __bf16 Hb [BM][CHN + 8];   // LN output, bf16 A-tile for fc2

  const int tid  = threadIdx.x;
  const int wid  = tid >> 5;
  const int lane = tid & 31;
  const int half = lane >> 4;
  const int l16  = lane & 15;
  const int tok0 = blockIdx.x * BM;

  // ---- phase A: router gates (fp32): 32 tokens x 4 preds, 2 lanes each ----
  if (npred > 0) {
    int sub = tid & 1, pair = tid >> 1;
    int t = pair >> 2, p = pair & 3;
    const float* xr = xnorm + (size_t)(tok0 + t) * DIM + sub * 512;
    const float* wr = rw + p * DIM + sub * 512;
    float acc = 0.f;
    for (int i = 0; i < 512; ++i) acc = fmaf(xr[i], wr[i], acc);
    acc += __shfl_xor(acc, 1, 32);
    if (sub == 0) wgt[t][p] = 1.f / (1.f + expf(-(acc + rb[p])));
  }
  __syncthreads();

  // ---- fc1: [32 x DIM] @ [DIM x CHN], split-K staged A, h kept in registers ----
  // wave owns 2 row tiles (tokens 0-15 / 16-31) x 2 column tiles
  v8f acc00 = {}, acc01 = {}, acc10 = {}, acc11 = {};
  const int n0 = (wid * 2 + 0) * 16 + l16;
  const int n1 = (wid * 2 + 1) * 16 + l16;
  const float* preds[4] = {p0, p1, p2, p3};

  for (int kh = 0; kh < 2; ++kh) {
    // fill A half: gated predecessor sum -> bf16
    for (int e = tid; e < BM * KH; e += 256) {
      int t = e >> 9, d0 = e & (KH - 1);
      size_t g = (size_t)(tok0 + t) * DIM + kh * KH + d0;
      float a;
      if (npred == 0) a = xnorm[g];
      else { a = 0.f; for (int p = 0; p < npred; ++p) a = fmaf(preds[p][g], wgt[t][p], a); }
      Ain[t][d0] = (__bf16)a;
    }
    __syncthreads();

    const __bf16* B0 = w1 + (size_t)n0 * DIM + kh * KH;
    const __bf16* B1 = w1 + (size_t)n1 * DIM + kh * KH;
    for (int k = 0; k < KH; k += 32) {
      int kA = k + half * 8;
      v16bf a0  = load_frag(&Ain[l16][kA]);
      v16bf a1  = load_frag(&Ain[16 + l16][kA]);
      v16bf bf0 = load_frag(B0 + kA);
      v16bf bf1 = load_frag(B1 + kA);
      acc00 = __builtin_amdgcn_wmma_f32_16x16x32_bf16(false, a0, false, bf0, (short)0, acc00, false, false);
      acc01 = __builtin_amdgcn_wmma_f32_16x16x32_bf16(false, a0, false, bf1, (short)0, acc01, false, false);
      acc10 = __builtin_amdgcn_wmma_f32_16x16x32_bf16(false, a1, false, bf0, (short)0, acc10, false, false);
      acc11 = __builtin_amdgcn_wmma_f32_16x16x32_bf16(false, a1, false, bf1, (short)0, acc11, false, false);
    }
    __syncthreads();   // Ain half will be overwritten next iteration
  }

  // ---- bias + exact gelu in registers, per-wave LN partial sums ----
  {
    float bias0 = b1[n0], bias1 = b1[n1];
#pragma unroll
    for (int i = 0; i < 8; ++i) {
      float v;
      v = acc00[i] + bias0; acc00[i] = 0.5f * v * (1.f + erff(v * 0.70710678118f));
      v = acc01[i] + bias1; acc01[i] = 0.5f * v * (1.f + erff(v * 0.70710678118f));
      v = acc10[i] + bias0; acc10[i] = 0.5f * v * (1.f + erff(v * 0.70710678118f));
      v = acc11[i] + bias1; acc11[i] = 0.5f * v * (1.f + erff(v * 0.70710678118f));
    }
#pragma unroll
    for (int i = 0; i < 8; ++i) {
      float s0 = acc00[i] + acc01[i];
      float q0 = acc00[i] * acc00[i] + acc01[i] * acc01[i];
      float s1 = acc10[i] + acc11[i];
      float q1 = acc10[i] * acc10[i] + acc11[i] * acc11[i];
#pragma unroll
      for (int m = 8; m >= 1; m >>= 1) {
        s0 += __shfl_xor(s0, m, 32); q0 += __shfl_xor(q0, m, 32);
        s1 += __shfl_xor(s1, m, 32); q1 += __shfl_xor(q1, m, 32);
      }
      if (l16 == 0) {
        int t = i + 8 * half;
        part[t][wid][0]      = s0; part[t][wid][1]      = q0;
        part[t + 16][wid][0] = s1; part[t + 16][wid][1] = q1;
      }
    }
  }
  __syncthreads();

  // ---- finalize LN stats: 32 tokens x 8 partials ----
  {
    int t = tid >> 3, q = tid & 7;
    float s = part[t][q][0], s2 = part[t][q][1];
#pragma unroll
    for (int m = 4; m >= 1; m >>= 1) { s += __shfl_xor(s, m, 32); s2 += __shfl_xor(s2, m, 32); }
    if (q == 0) {
      float mu  = s  * (1.f / CHN);
      float var = s2 * (1.f / CHN) - mu * mu;
      musr[t][0] = mu;
      musr[t][1] = rsqrtf(var + 1e-5f);
    }
  }
  __syncthreads();

  // ---- normalize register h -> bf16 fc2 A-tile ----
  {
    float g0 = lng[n0], g1 = lng[n1], be0 = lnb[n0], be1 = lnb[n1];
#pragma unroll
    for (int i = 0; i < 8; ++i) {
      int ta = i + 8 * half, tb = ta + 16;
      float mua = musr[ta][0], ra = musr[ta][1];
      float mub = musr[tb][0], rbs = musr[tb][1];
      Hb[ta][n0] = (__bf16)((acc00[i] - mua) * ra  * g0 + be0);
      Hb[ta][n1] = (__bf16)((acc01[i] - mua) * ra  * g1 + be1);
      Hb[tb][n0] = (__bf16)((acc10[i] - mub) * rbs * g0 + be0);
      Hb[tb][n1] = (__bf16)((acc11[i] - mub) * rbs * g1 + be1);
    }
  }
  __syncthreads();

  // ---- fc2: [32 x CHN] @ [CHN x DIM]; each weight row read once per block ----
  for (int g = 0; g < 2; ++g) {
    v8f zero = {};
    v8f acc[2][4];
#pragma unroll
    for (int c = 0; c < 4; ++c) { acc[0][c] = zero; acc[1][c] = zero; }
    const int cb = wid * 8 + g * 4;   // column-tile base (of 64)
    for (int k = 0; k < CHN; k += 32) {
      int kA = k + half * 8;
      v16bf a0 = load_frag(&Hb[l16][kA]);
      v16bf a1 = load_frag(&Hb[16 + l16][kA]);
#pragma unroll
      for (int c = 0; c < 4; ++c) {
        int n = (cb + c) * 16 + l16;
        v16bf bf = load_frag(w2 + (size_t)n * CHN + kA);
        acc[0][c] = __builtin_amdgcn_wmma_f32_16x16x32_bf16(false, a0, false, bf, (short)0, acc[0][c], false, false);
        acc[1][c] = __builtin_amdgcn_wmma_f32_16x16x32_bf16(false, a1, false, bf, (short)0, acc[1][c], false, false);
      }
    }
#pragma unroll
    for (int c = 0; c < 4; ++c) {
      int n = cb * 16 + c * 16 + l16;
#pragma unroll
      for (int i = 0; i < 8; ++i) {
        int ta = i + 8 * half, tb = ta + 16;
        size_t ga = (size_t)(tok0 + ta) * DIM + n;
        size_t gb = (size_t)(tok0 + tb) * DIM + n;
        if (terminal) {
          float fb = fc2_bias[n];
          outbuf[ga] = x[ga] + acc[0][c][i] + fb;
          outbuf[gb] = x[gb] + acc[1][c][i] + fb;
        } else {
          outbuf[ga] = acc[0][c][i];
          outbuf[gb] = acc[1][c][i];
        }
      }
    }
  }
}

// ---------------- host launcher ----------------
extern "C" void kernel_launch(void* const* d_in, const int* in_sizes, int n_in,
                              void* d_out, int out_size, void* d_ws, size_t ws_size,
                              hipStream_t stream) {
  const float* x        = (const float*)d_in[0];
  const float* norm_g   = (const float*)d_in[1];
  const float* norm_b   = (const float*)d_in[2];
  const float* fc1_w    = (const float*)d_in[3];
  const float* fc1_b    = (const float*)d_in[4];
  const float* ln_g     = (const float*)d_in[5];
  const float* ln_b     = (const float*)d_in[6];
  const float* fc2_w    = (const float*)d_in[7];
  const float* fc2_bias = (const float*)d_in[8];
  const float* router_w = (const float*)d_in[9];
  const float* router_b = (const float*)d_in[10];
  float* out = (float*)d_out;

  // workspace carve-up (~251 MB): bf16 weights, x_norm, 6 rotating node-output slots
  char* ws = (char*)d_ws;
  size_t off = 0;
  auto take = [&](size_t bytes) -> char* {
    char* p = ws + off; off = (off + bytes + 255) & ~(size_t)255; return p;
  };
  __bf16* w1b   = (__bf16*)take((size_t)16 * CHN * DIM * 2);
  __bf16* w2b   = (__bf16*)take((size_t)16 * DIM * CHN * 2);
  float*  xn    = (float*) take((size_t)TOK * DIM * 4);
  float*  slots = (float*) take((size_t)6 * TOK * DIM * 4);

  cvt_bf16_kernel<<<4096, 256, 0, stream>>>(fc1_w, w1b, 16 * CHN * DIM);
  cvt_bf16_kernel<<<4096, 256, 0, stream>>>(fc2_w, w2b, 16 * DIM * CHN);
  xnorm_kernel<<<TOK / 8, 256, 0, stream>>>(x, norm_g, norm_b, xn);

  // Watts-Strogatz ring-lattice DAG (matches make_preds(16))
  static const int NP[16] = {0,1,2,2,2,2,2,2,2,2,2,2,2,2,3,4};
  static const int PL[16][4] = {
    {0,0,0,0},{0,0,0,0},{0,1,0,0},{1,2,0,0},{2,3,0,0},{3,4,0,0},
    {4,5,0,0},{5,6,0,0},{6,7,0,0},{7,8,0,0},{8,9,0,0},{9,10,0,0},
    {10,11,0,0},{11,12,0,0},{0,12,13,0},{0,1,13,14}};
  // slot reuse: keep 0 and 1 pinned (read by nodes 14/15), rotate 4 slots for the rest
  auto slotptr = [&](int v) -> float* {
    int s = (v == 0) ? 0 : (v == 1) ? 1 : (2 + ((v - 2) & 3));
    return slots + (size_t)s * ((size_t)TOK * DIM);
  };

  for (int v = 0; v < 16; ++v) {
    const float* p[4];
    for (int i = 0; i < 4; ++i) p[i] = (i < NP[v]) ? slotptr(PL[v][i]) : xn;  // dummy if unused
    float* ob = (v == 15) ? out : slotptr(v);
    node_kernel<<<TOK / BM, 256, 0, stream>>>(
        xn,
        w1b + (size_t)v * CHN * DIM, fc1_b + v * CHN,
        ln_g + v * CHN, ln_b + v * CHN,
        w2b + (size_t)v * DIM * CHN,
        router_w + v * 4 * DIM, router_b + v * 4,
        p[0], p[1], p[2], p[3], NP[v],
        ob, x, fc2_bias, (v == 15) ? 1 : 0);
  }
  (void)in_sizes; (void)n_in; (void)out_size; (void)ws_size;
}
